// QGenBelief_76879914598604
// MI455X (gfx1250) — matrix operand order repaired and checked by reading, not compiled, not run on hardware
//
#include <hip/hip_runtime.h>
#include <hip/hip_bf16.h>

// Problem constants (match reference harness).
#define BB 64
#define QQ 16
#define OO 20
#define DD 256
#define TT 4096

typedef __attribute__((ext_vector_type(16))) _Float16 v16h;
typedef __attribute__((ext_vector_type(8)))  float    v8f;
typedef __attribute__((ext_vector_type(4)))  float    f32x4;

// gfx1250 hardware tanh (TRANS32 op). v_nop after per ISA trans-op hazard rule,
// since the compiler can't schedule around inline asm.
__device__ __forceinline__ float tanh_hw(float x) {
    float y;
    asm volatile("v_tanh_f32 %0, %1\n\tv_nop" : "=v"(y) : "v"(x));
    return y;
}

// ---------------------------------------------------------------------------
// Kernel 1: belief = tanh( softmax(logits) @ emb[cats] )   -> [B, Q, D] f32
// One wave per (batch, d-tile of 16). 64*16 = 1024 waves, 1 WMMA each.
// A fragment (16x32 f16): probs padded K=20 -> 32 with zeros.
// B fragment (32x16 f16): gathered embedding columns, K>=20 zero.
// ---------------------------------------------------------------------------
__global__ __launch_bounds__(256) void belief_wmma_kernel(
    const float* __restrict__ logits,   // [B, Q, O]
    const float* __restrict__ emb,      // [NUM_CAT, D]
    const int*   __restrict__ cats,     // [B, O]
    float*       __restrict__ belief)   // [B, Q, D]
{
    const int wave = (blockIdx.x * blockDim.x + threadIdx.x) >> 5;
    const int lane = threadIdx.x & 31;
    const int b    = wave >> 4;        // D/16 = 16 tiles per batch
    const int dt   = wave & 15;
    const int m    = lane & 15;        // A: row (question). B: column N.
    const int hi   = lane >> 4;        // lane half (wave32 layout split)

    // --- softmax of row m of logits[b] (O=20), computed in f32 ---
    float lrow[OO];
    float mx = -INFINITY;
#pragma unroll
    for (int o = 0; o < OO; ++o) {
        lrow[o] = logits[(b * QQ + m) * OO + o];
        mx = fmaxf(mx, lrow[o]);
    }
    float sum = 0.0f;
#pragma unroll
    for (int o = 0; o < OO; ++o) {
        lrow[o] = __expf(lrow[o] - mx);
        sum += lrow[o];
    }
    const float inv = 1.0f / sum;

    // --- A fragment: 16-bit A 16x32 layout (ISA 7.12.2) ---
    // lane<16: halves 0..7 -> K=0..7,   halves 8..15 -> K=16..23
    // lane>=16: halves 0..7 -> K=8..15, halves 8..15 -> K=24..31
    v16h a;
#pragma unroll
    for (int j = 0; j < 16; ++j) {
        const int K = (j < 8 ? j : j + 8) + hi * 8;
        const float v = (K < OO) ? lrow[K] * inv : 0.0f;
        a[j] = (_Float16)v;
    }

    // --- B fragment: 16-bit B 32x16 layout ---
    // lane<16: halves 0..7 -> K=0..7, 8..15 -> K=16..23; lane>=16: K=8..15, K=24..31
    const int col = dt * 16 + m;       // global embedding dim
    v16h bm;
#pragma unroll
    for (int j = 0; j < 16; ++j) {
        const int K = (j < 8 ? j : j + 8) + hi * 8;
        float v = 0.0f;
        if (K < OO) {
            const int cat = cats[b * OO + K];
            v = emb[cat * DD + col];
        }
        bm[j] = (_Float16)v;
    }

    v8f c = {};
    c = __builtin_amdgcn_wmma_f32_16x16x32_f16(
            /*neg_a=*/false, a, /*neg_b=*/false, bm,
            /*c_mod=*/(short)0, c, /*reuse_a=*/false, /*reuse_b=*/false);

    // --- store D: VGPR r, lane<16 -> M=r ; lane>=16 -> M=r+8 ; N=lane&15 ---
#pragma unroll
    for (int r = 0; r < 8; ++r) {
        const int M = r + hi * 8;
        belief[(b * QQ + M) * DD + col] = tanh_hw(c[r]);
    }
}

// ---------------------------------------------------------------------------
// Kernel 2: scatter belief rows to tokens. 256 MB of streaming f32 output:
// pure store-bandwidth bound -> float4 (b128) non-temporal stores so the
// write stream (256 MB > 192 MB L2) does not evict the hot 1 MB belief table.
// Block = 256 threads = 4 tokens x 64 float4 lanes. Boundary data is
// block-uniform -> scalar loads.
// ---------------------------------------------------------------------------
__global__ __launch_bounds__(256) void scatter_tokens_kernel(
    const float* __restrict__ belief,   // [B, Q, D]
    const int*   __restrict__ cum,      // [B, Q]
    const int*   __restrict__ nq,       // [B]
    float*       __restrict__ out)      // [B, T, D]
{
    const int blocksPerBatch = TT / 4;                 // 1024
    const int b     = blockIdx.x / blocksPerBatch;
    const int tbase = (blockIdx.x % blocksPerBatch) * 4;
    const int tl    = threadIdx.x >> 6;                // token within block
    const int d4    = threadIdx.x & 63;                // float4 index in D
    const int t     = tbase + tl;

    const int nqb = nq[b];

    // qi = number of boundaries <= t; boundary j valid iff j <= nqb-1
    int qi = 0;
#pragma unroll
    for (int j = 1; j < QQ; ++j) {
        const int bnd = (j <= nqb - 1) ? (cum[b * QQ + j] - 1) : (TT + 1);
        qi += (t >= bnd) ? 1 : 0;
    }
    const bool valid = (qi < nqb - 1);
    const int  qc    = qi < (QQ - 1) ? qi : (QQ - 1);

    f32x4 v = {0.0f, 0.0f, 0.0f, 0.0f};
    if (valid) {
        const f32x4* src = (const f32x4*)(belief + (size_t)(b * QQ + qc) * DD);
        v = src[d4];
    }
    f32x4* dst = (f32x4*)(out + ((size_t)b * TT + t) * DD);
    __builtin_nontemporal_store(v, dst + d4);
}

// ---------------------------------------------------------------------------
extern "C" void kernel_launch(void* const* d_in, const int* in_sizes, int n_in,
                              void* d_out, int out_size, void* d_ws, size_t ws_size,
                              hipStream_t stream) {
    const float* logits = (const float*)d_in[0];   // [B,Q,O] f32
    const float* emb    = (const float*)d_in[1];   // [NUM_CAT,D] f32
    const int*   cats   = (const int*)  d_in[2];   // [B,O]
    const int*   cum    = (const int*)  d_in[3];   // [B,Q]
    const int*   nq     = (const int*)  d_in[4];   // [B]
    // d_in[5] = max_dln scalar (compile-time TT here)

    float* belief = (float*)d_ws;                  // [B,Q,D] = 1 MB scratch
    float* out    = (float*)d_out;                 // [B,T,D]

    // Kernel 1: 1024 waves (B * D/16), 8 waves per 256-thread block.
    const int waves1   = BB * (DD / 16);
    const int threads1 = waves1 * 32;
    belief_wmma_kernel<<<threads1 / 256, 256, 0, stream>>>(logits, emb, cats, belief);

    // Kernel 2: one float4 NT store per thread; B * T/4 blocks of 256.
    scatter_tokens_kernel<<<BB * (TT / 4), 256, 0, stream>>>(belief, cum, nq, out);
}
